// BLSTMCell_33423435498265
// MI455X (gfx1250) — matrix operand order repaired and checked by reading, not compiled, not run on hardware
//
#include <hip/hip_runtime.h>
#include <hip/hip_bf16.h>

typedef _Float16 v8h  __attribute__((ext_vector_type(8)));
typedef _Float16 v16h __attribute__((ext_vector_type(16)));
typedef float    v8f  __attribute__((ext_vector_type(8)));
typedef int      v4i  __attribute__((vector_size(4 * sizeof(int))));

#define B_SZ 8192
#define IN_SZ 512
#define H_SZ 512
#define K_SZ 1024   // IN + H (concatenated GEMM K)
#define G_SZ 2048   // 4*H gate columns
#define BM   256    // rows per block (8 waves * 32 rows)
#define KC   64     // K-chunk staged in LDS (2 WMMA k-steps)

#if __has_builtin(__builtin_amdgcn_global_load_async_to_lds_b128)
#define HAS_ASYNC_LDS 1
typedef __attribute__((address_space(1))) v4i gv4i;   // global v4i32
typedef __attribute__((address_space(3))) v4i lv4i;   // LDS v4i32
#else
#define HAS_ASYNC_LDS 0
#endif

#if __has_builtin(__builtin_amdgcn_s_wait_asynccnt)
#define WAIT_ASYNC(n) __builtin_amdgcn_s_wait_asynccnt(n)
#else
#define WAIT_ASYNC(n) asm volatile("s_wait_asynccnt %0" ::"i"(n) : "memory")
#endif

// ---------------- prep kernels ----------------

// A[b][k] = f16( k<512 ? x[b][k] : hx[b][k-512] )
__global__ void pack_A_kernel(const float* __restrict__ x, const float* __restrict__ hx,
                              _Float16* __restrict__ A) {
  int idx = blockIdx.x * blockDim.x + threadIdx.x;       // 0 .. B*K-1
  int b = idx >> 10, k = idx & 1023;
  float v = (k < IN_SZ) ? x[(size_t)b * IN_SZ + k] : hx[(size_t)b * H_SZ + (k - IN_SZ)];
  A[idx] = (_Float16)v;
}

// Wc[g][k] = f16( sign( k<512 ? W_ih[g][k] : W_hh[g][k-512] ) ), sign(0)=0 (jnp.sign)
__global__ void pack_W_kernel(const float* __restrict__ Wih, const float* __restrict__ Whh,
                              _Float16* __restrict__ Wc) {
  int idx = blockIdx.x * blockDim.x + threadIdx.x;       // 0 .. G*K-1
  int g = idx >> 10, k = idx & 1023;
  float w = (k < IN_SZ) ? Wih[(size_t)g * IN_SZ + k] : Whh[(size_t)g * H_SZ + (k - IN_SZ)];
  float s = (w > 0.0f) ? 1.0f : ((w < 0.0f) ? -1.0f : 0.0f);
  Wc[idx] = (_Float16)s;
}

__global__ void pack_bias_kernel(const float* __restrict__ bih, const float* __restrict__ bhh,
                                 float* __restrict__ bsum) {
  int i = blockIdx.x * blockDim.x + threadIdx.x;
  bsum[i] = bih[i] + bhh[i];
}

// ---------------- main fused GEMM + LSTM epilogue ----------------

__device__ __forceinline__ float sigmoidf_(float v) { return 1.0f / (1.0f + __expf(-v)); }

__global__ __launch_bounds__(256)
void blstm_wmma_kernel(const _Float16* __restrict__ A,    // [B_SZ][K_SZ] f16
                       const _Float16* __restrict__ Wc,   // [G_SZ][K_SZ] f16 (+/-1)
                       const float* __restrict__ bsum,    // [G_SZ]
                       const float* __restrict__ cx,      // [B_SZ][H_SZ]
                       float* __restrict__ hy,            // [B_SZ][H_SZ]
                       float* __restrict__ cy) {          // [B_SZ][H_SZ]
  __shared__ _Float16 sA[2][BM * 64];   // 2 x 32 KB  [row][k]
  __shared__ _Float16 sB[2][64 * 64];   // 2 x  8 KB  [gatecol][k]

  const int n0   = blockIdx.x * 16;   // h-tile base
  const int m0   = blockIdx.y * BM;   // row-tile base
  const int tid  = threadIdx.x;
  const int wave = tid >> 5;
  const int lane = tid & 31;

  v8f acc[2][4] = {};                 // [m-subtile][gate]

  // Stage one K-chunk (A: 8 b128/thread, B: 2 b128/thread -> 10 async ops/wave).
  auto stage = [&](int buf, int kc) {
#if HAS_ASYNC_LDS
    #pragma unroll
    for (int i = 0; i < 8; ++i) {
      int c = tid + 256 * i;          // 2048 v8h chunks for A tile
      int row = c >> 3;
      int kin = (c & 7) * 8;
      __builtin_amdgcn_global_load_async_to_lds_b128(
          (gv4i*)&A[(size_t)(m0 + row) * K_SZ + kc + kin],
          (lv4i*)&sA[buf][row * 64 + kin], 0, 0);
    }
    #pragma unroll
    for (int i = 0; i < 2; ++i) {
      int c = tid + 256 * i;          // 512 v8h chunks for B tile
      int j = c >> 3;
      int kin = (c & 7) * 8;
      int gc = (j >> 4) * H_SZ + n0 + (j & 15);
      __builtin_amdgcn_global_load_async_to_lds_b128(
          (gv4i*)&Wc[(size_t)gc * K_SZ + kc + kin],
          (lv4i*)&sB[buf][j * 64 + kin], 0, 0);
    }
#else
    #pragma unroll
    for (int i = 0; i < 8; ++i) {
      int c = tid + 256 * i;
      int row = c >> 3;
      int kin = (c & 7) * 8;
      *(v8h*)&sA[buf][row * 64 + kin] =
          *(const v8h*)&A[(size_t)(m0 + row) * K_SZ + kc + kin];
    }
    #pragma unroll
    for (int i = 0; i < 2; ++i) {
      int c = tid + 256 * i;
      int j = c >> 3;
      int kin = (c & 7) * 8;
      int gc = (j >> 4) * H_SZ + n0 + (j & 15);
      *(v8h*)&sB[buf][j * 64 + kin] =
          *(const v8h*)&Wc[(size_t)gc * K_SZ + kc + kin];
    }
#endif
  };

  stage(0, 0);                        // prologue: fill buffer 0

  for (int it = 0; it < K_SZ / KC; ++it) {
    const int buf = it & 1;
    const bool hasNext = (it + 1) < (K_SZ / KC);

    __syncthreads();                  // everyone done reading buffer buf^1
    if (hasNext) stage(buf ^ 1, (it + 1) * KC);
#if HAS_ASYNC_LDS
    if (hasNext) { WAIT_ASYNC(10); }  // current buffer landed; next 10 in flight
    else         { WAIT_ASYNC(0);  }
#endif
    __syncthreads();                  // current buffer visible to all waves

    #pragma unroll
    for (int ks = 0; ks < KC; ks += 32) {
      // ---- preload all fragments, then burst 8 WMMAs ----
      // A fragment per CDNA5 16-bit A 16x32 layout:
      //  lane<16:  halfs 0..7 = K 0..7,  halfs 8..15 = K 16..23 (row = lane)
      //  lane>=16: halfs 0..7 = K 8..15, halfs 8..15 = K 24..31 (row = lane-16)
      const int ar = lane & 15;
      const int kb = (lane >> 4) * 8;
      v16h af[2];
      #pragma unroll
      for (int t = 0; t < 2; ++t) {
        const _Float16* base = &sA[buf][(wave * 32 + t * 16 + ar) * 64 + ks + kb];
        v8h lo = *(const v8h*)base;
        v8h hi = *(const v8h*)(base + 16);
        af[t] = __builtin_shufflevector(lo, hi,
            0,1,2,3,4,5,6,7,8,9,10,11,12,13,14,15);
      }
      // B fragment: col=lane&15, same K layout as A (halfs 0..7 = K kb..kb+7, 8..15 = K 16+kb..).
      const int kb2 = (lane >> 4) * 8;
      v16h bf[4];
      #pragma unroll
      for (int g = 0; g < 4; ++g) {
        const _Float16* bb = &sB[buf][(g * 16 + (lane & 15)) * 64 + ks + kb2];
        v8h blo = *(const v8h*)bb;
        v8h bhi = *(const v8h*)(bb + 16);
        bf[g] = __builtin_shufflevector(blo, bhi, 0,1,2,3,4,5,6,7,8,9,10,11,12,13,14,15);
      }
      #pragma unroll
      for (int g = 0; g < 4; ++g) {
        acc[0][g] = __builtin_amdgcn_wmma_f32_16x16x32_f16(
            false, af[0], false, bf[g], (short)0, acc[0][g], false, false);
        acc[1][g] = __builtin_amdgcn_wmma_f32_16x16x32_f16(
            false, af[1], false, bf[g], (short)0, acc[1][g], false, false);
      }
    }
  }

  // ---- fused LSTM epilogue ----
  // C/D layout: lane<16 -> N=lane, elem r -> M=r; lane>=16 -> N=lane-16, M=8+r.
  const int n  = lane & 15;
  const int mb = (lane >> 4) * 8;
  const int h  = n0 + n;
  const float bi = bsum[h];
  const float bf_ = bsum[h + 512];
  const float bg = bsum[h + 1024];
  const float bo = bsum[h + 1536];

  #pragma unroll
  for (int t = 0; t < 2; ++t) {
    #pragma unroll
    for (int r = 0; r < 8; ++r) {
      const int m = m0 + wave * 32 + t * 16 + mb + r;
      float ig = acc[t][0][r] + bi;
      float fg = acc[t][1][r] + bf_;
      float cg = acc[t][2][r] + bg;
      float og = acc[t][3][r] + bo;
      float c_old = cx[(size_t)m * H_SZ + h];
      float cnew  = sigmoidf_(fg) * c_old + sigmoidf_(ig) * tanhf(cg);
      cy[(size_t)m * H_SZ + h] = cnew;
      hy[(size_t)m * H_SZ + h] = sigmoidf_(og) * tanhf(cnew);
    }
  }
}

// ---------------- launch ----------------

extern "C" void kernel_launch(void* const* d_in, const int* in_sizes, int n_in,
                              void* d_out, int out_size, void* d_ws, size_t ws_size,
                              hipStream_t stream) {
  const float* x   = (const float*)d_in[0];
  const float* hx  = (const float*)d_in[1];
  const float* cx  = (const float*)d_in[2];
  const float* Wih = (const float*)d_in[3];
  const float* Whh = (const float*)d_in[4];
  const float* bih = (const float*)d_in[5];
  const float* bhh = (const float*)d_in[6];

  char* ws = (char*)d_ws;
  _Float16* A  = (_Float16*)ws;                                   // 16 MB
  _Float16* Wc = (_Float16*)(ws + (size_t)B_SZ * K_SZ * 2);       //  4 MB
  float*  bsum = (float*)(ws + (size_t)B_SZ * K_SZ * 2
                             + (size_t)G_SZ * K_SZ * 2);          //  8 KB

  float* hy  = (float*)d_out;
  float* cyp = hy + (size_t)B_SZ * H_SZ;

  pack_A_kernel<<<(B_SZ * K_SZ) / 256, 256, 0, stream>>>(x, hx, A);
  pack_W_kernel<<<(G_SZ * K_SZ) / 256, 256, 0, stream>>>(Wih, Whh, Wc);
  pack_bias_kernel<<<G_SZ / 256, 256, 0, stream>>>(bih, bhh, bsum);

  dim3 grid(H_SZ / 16, B_SZ / BM);   // 32 x 32 blocks
  blstm_wmma_kernel<<<grid, 256, 0, stream>>>(A, Wc, bsum, cx, hy, cyp);
}
